// ProgInfNet_927712936072
// MI455X (gfx1250) — matrix operand order repaired and checked by reading, not compiled, not run on hardware
//
#include <hip/hip_runtime.h>
#include <hip/hip_bf16.h>

// ---------------- model constants (match reference) ----------------
constexpr int kHD = 256, kNL = 4, kNH = 8, kDH = 32;
constexpr int kMS = 32, kMP = 8, kT = 40, kES = 64;
constexpr int kBATCH = 8, kBEAMS = 5, kBB = 40;
constexpr float kMINLL = -100000.0f;
constexpr int kSTART = 1, kEND = 2;
constexpr int kHROWS = 64;   // head rows padded 40 -> 64 (multiple of 32)

typedef __attribute__((ext_vector_type(16))) _Float16 v16h;
typedef __attribute__((ext_vector_type(8)))  float    v8f;

// ---------------- WMMA fragment loaders (CDNA5 ISA 7.12.2) ----
union FragH { v16h v; unsigned u[8]; };

// A: row-major M x K f16, 16x32 tile. lane m = lane&15.
// lanes 0-15: halves 0..7 -> K 0..7, halves 8..15 -> K 16..23
// lanes 16-31: halves 0..7 -> K 8..15, halves 8..15 -> K 24..31
__device__ inline v16h load_a16x32(const _Float16* A, int lda, int m0, int k0, int lane) {
  FragH f;
  const _Float16* p = A + (size_t)(m0 + (lane & 15)) * lda + k0 + ((lane >> 4) << 3);
#pragma unroll
  for (int j = 0; j < 4; ++j) {
    f.u[j]     = *(const unsigned*)(p + 2 * j);        // K = 2j,2j+1 (+8 hi)
    f.u[j + 4] = *(const unsigned*)(p + 16 + 2 * j);   // K = 16+2j   (+8 hi)
  }
  return f.v;
}

// B: stored transposed (N x K row-major), 32x16 tile. lane n = lane&15.
// lanes 0-15: halves 0..7 -> K 0..7, halves 8..15 -> K 16..23 ; lanes 16-31: K 8..15, K 24..31
__device__ inline v16h load_b32x16T(const _Float16* Bt, int ldb, int n0, int k0, int lane) {
  FragH f;
  const _Float16* p = Bt + (size_t)(n0 + (lane & 15)) * ldb + k0 + ((lane >> 4) << 3);
#pragma unroll
  for (int j = 0; j < 8; ++j) f.u[j] = *(const unsigned*)(p + 2 * (j & 3) + ((j >> 2) << 4));
  return f.v;
}

// epilogue for one 16x16 C tile
template<bool BIAS, bool RELU, bool RESID, bool OUTF16>
__device__ inline void store_tile(v8f acc, int m0, int n0, int lane,
                                  const float* __restrict__ bias,
                                  const float* __restrict__ resid,
                                  float* __restrict__ Cf, _Float16* __restrict__ Ch, int N) {
  const int n  = n0 + (lane & 15);
  const int mo = m0 + ((lane >> 4) << 3);   // C/D: vgpr j -> m = j + 8*(lane>=16)
  float bv = BIAS ? bias[n] : 0.0f;
#pragma unroll
  for (int j = 0; j < 8; ++j) {
    float v = acc[j] + bv;
    if (RESID) v += resid[(size_t)(mo + j) * N + n];
    if (RELU)  v = v > 0.0f ? v : 0.0f;
    if (OUTF16) Ch[(size_t)(mo + j) * N + n] = (_Float16)v;
    else        Cf[(size_t)(mo + j) * N + n] = v;
  }
}

// ---------------- WMMA GEMM: C(MxN) = A(MxK) * Bt(NxK)^T (+bias,+resid,relu) ---
// one wave32 per 32x32 output tile (2x2 register blocking -> 4 WMMAs / k-step,
// halves b128 load traffic per WMMA vs 16x16 tiles);
// grid = (N/32, M/32), block = 32
template<bool BIAS, bool RELU, bool RESID, bool OUTF16>
__global__ void gemm_wmma_kernel(const _Float16* __restrict__ A,
                                 const _Float16* __restrict__ Bt,
                                 const float* __restrict__ bias,
                                 const float* __restrict__ resid,
                                 float* __restrict__ Cf, _Float16* __restrict__ Ch,
                                 int M, int N, int K) {
  const int lane = threadIdx.x;
  const int n0 = blockIdx.x * 32;
  const int m0 = blockIdx.y * 32;
  v8f acc00 = {}, acc01 = {}, acc10 = {}, acc11 = {};
  for (int k0 = 0; k0 < K; k0 += 32) {
    v16h a0 = load_a16x32(A, K, m0,      k0, lane);
    v16h a1 = load_a16x32(A, K, m0 + 16, k0, lane);
    v16h b0 = load_b32x16T(Bt, K, n0,      k0, lane);
    v16h b1 = load_b32x16T(Bt, K, n0 + 16, k0, lane);
    acc00 = __builtin_amdgcn_wmma_f32_16x16x32_f16(false, a0, false, b0, (short)0, acc00, false, false);
    acc01 = __builtin_amdgcn_wmma_f32_16x16x32_f16(false, a0, false, b1, (short)0, acc01, false, false);
    acc10 = __builtin_amdgcn_wmma_f32_16x16x32_f16(false, a1, false, b0, (short)0, acc10, false, false);
    acc11 = __builtin_amdgcn_wmma_f32_16x16x32_f16(false, a1, false, b1, (short)0, acc11, false, false);
  }
  store_tile<BIAS, RELU, RESID, OUTF16>(acc00, m0,      n0,      lane, bias, resid, Cf, Ch, N);
  store_tile<BIAS, RELU, RESID, OUTF16>(acc01, m0,      n0 + 16, lane, bias, resid, Cf, Ch, N);
  store_tile<BIAS, RELU, RESID, OUTF16>(acc10, m0 + 16, n0,      lane, bias, resid, Cf, Ch, N);
  store_tile<BIAS, RELU, RESID, OUTF16>(acc11, m0 + 16, n0 + 16, lane, bias, resid, Cf, Ch, N);
}

// ---------------- weight convert + transpose f32(KxN) -> f16(NxK) ----------
__global__ void wt_f16_kernel(const float* __restrict__ W, _Float16* __restrict__ Wt,
                              int K, int N) {
  int l = blockIdx.z;
  int idx = blockIdx.x * blockDim.x + threadIdx.x;
  if (idx >= K * N) return;
  int k = idx % K, n = idx / K;
  Wt[(size_t)l * K * N + idx] = (_Float16)W[(size_t)l * K * N + (size_t)k * N + n];
}

// ---------------- LayerNorm (f32 -> f16), one wave per 256-wide row ---------
__global__ void ln_kernel(const float* __restrict__ x, const float* __restrict__ s,
                          const float* __restrict__ b, _Float16* __restrict__ h, int rows) {
  int wave = threadIdx.x >> 5;
  int lane = threadIdx.x & 31;
  int row = blockIdx.x * 8 + wave;
  if (row >= rows) return;
  const float* xr = x + (size_t)row * kHD;
  float vals[8], sum = 0.f, sq = 0.f;
#pragma unroll
  for (int i = 0; i < 8; ++i) {
    float v = xr[lane + 32 * i];
    vals[i] = v; sum += v; sq += v * v;
  }
#pragma unroll
  for (int o = 16; o > 0; o >>= 1) {
    sum += __shfl_xor(sum, o, 32);
    sq  += __shfl_xor(sq,  o, 32);
  }
  float mean = sum * (1.0f / kHD);
  float var  = sq  * (1.0f / kHD) - mean * mean;
  float inv  = rsqrtf(var + 1e-5f);
  _Float16* hr = h + (size_t)row * kHD;
#pragma unroll
  for (int i = 0; i < 8; ++i) {
    int c = lane + 32 * i;
    hr[c] = (_Float16)((vals[i] - mean) * inv * s[c] + b[c]);
  }
}

// ---------------- embedding: codes | tok_emb[seq]  + pos ------------------
__global__ void embed_kernel(const float* __restrict__ bcodes, const float* __restrict__ tok,
                             const float* __restrict__ pos, const int* __restrict__ bseqs,
                             float* __restrict__ x) {
  int idx = blockIdx.x * blockDim.x + threadIdx.x;
  if (idx >= kBB * kT * kHD) return;
  int c = idx % kHD;
  int t = (idx / kHD) % kT;
  int s = idx / (kHD * kT);
  float v;
  if (t < kMP) v = bcodes[((size_t)s * kMP + t) * kHD + c];
  else         v = tok[(size_t)bseqs[s * kMS + (t - kMP)] * kHD + c];
  x[idx] = v + pos[t * kHD + c];
}

// ---------------- causal attention, one block per (seq, head) -------------
__global__ void attn_kernel(const float* __restrict__ qkv, _Float16* __restrict__ out) {
  int s  = blockIdx.x / kNH;
  int hh = blockIdx.x % kNH;
  int t  = threadIdx.x;            // blockDim = 64, threads >= kT idle
  if (t >= kT) return;
  const float* qrow = qkv + (size_t)(s * kT + t) * 768 + hh * kDH;
  float scores[kT];
  float mx = -1e30f;
  for (int k = 0; k <= t; ++k) {
    const float* krow = qkv + (size_t)(s * kT + k) * 768 + 256 + hh * kDH;
    float d = 0.f;
#pragma unroll
    for (int c = 0; c < kDH; ++c) d += qrow[c] * krow[c];
    d *= 0.17677669529663687f;     // 1/sqrt(32)
    scores[k] = d;
    mx = fmaxf(mx, d);
  }
  float ssum = 0.f;
  for (int k = 0; k <= t; ++k) { scores[k] = __expf(scores[k] - mx); ssum += scores[k]; }
  float rs = 1.f / ssum;
  float acc[kDH];
#pragma unroll
  for (int c = 0; c < kDH; ++c) acc[c] = 0.f;
  for (int k = 0; k <= t; ++k) {
    const float* vrow = qkv + (size_t)(s * kT + k) * 768 + 512 + hh * kDH;
    float w = scores[k] * rs;
#pragma unroll
    for (int c = 0; c < kDH; ++c) acc[c] += w * vrow[c];
  }
  _Float16* orow = out + (size_t)(s * kT + t) * kHD + hh * kDH;
#pragma unroll
  for (int c = 0; c < kDH; ++c) orow[c] = (_Float16)acc[c];
}

// ---------------- gather head-input rows (position MP+ti) into 64x256 f16 --
__global__ void headgather_kernel(const float* __restrict__ x, _Float16* __restrict__ hin,
                                  int ti) {
  int idx = blockIdx.x * blockDim.x + threadIdx.x;
  if (idx >= kHROWS * kHD) return;
  int c = idx % kHD, r = idx / kHD;
  float v = 0.f;
  if (r < kBB) v = x[((size_t)r * kT + kMP + ti) * kHD + c];
  hin[idx] = (_Float16)v;
}

// ---------------- beam step: log-softmax, per-beam top5, merge, reorder ----
__global__ void beam_kernel(const float* __restrict__ logits, const int* __restrict__ seq_in,
                            int* __restrict__ seq_out, float* __restrict__ blls, int ti) {
  __shared__ float dist[kBEAMS][kES];
  __shared__ float liks[kBEAMS * kBEAMS];
  __shared__ int   choice[kBEAMS * kBEAMS];
  __shared__ float oldll[kBEAMS];
  __shared__ int   ranked[kBEAMS];
  __shared__ float rankedval[kBEAMS];
  int b = blockIdx.x;
  int tid = threadIdx.x;
  if (tid < kBEAMS) {
    const float* lg = logits + (size_t)(b * kBEAMS + tid) * kES;
    float mx = lg[0];
    for (int i = 1; i < kES; ++i) mx = fmaxf(mx, lg[i]);
    float s = 0.f;
    for (int i = 0; i < kES; ++i) s += __expf(lg[i] - mx);
    float inv = 1.f / s;
    for (int i = 0; i < kES; ++i) dist[tid][i] = __logf(__expf(lg[i] - mx) * inv + 1e-8f);
    oldll[tid] = blls[b * kBEAMS + tid];
    for (int j = 0; j < kBEAMS; ++j) {          // top-5, ties -> lowest index
      float best = -3.0e38f; int bi = 0;
      for (int i = 0; i < kES; ++i)
        if (dist[tid][i] > best) { best = dist[tid][i]; bi = i; }
      liks[tid * kBEAMS + j]   = best + oldll[tid];
      choice[tid * kBEAMS + j] = bi;
      dist[tid][bi] = -3.0e38f;
    }
  }
  __syncthreads();
  if (tid == 0) {                               // stable argsort-desc, take top 5 of 25
    bool used[kBEAMS * kBEAMS];
    for (int i = 0; i < kBEAMS * kBEAMS; ++i) used[i] = false;
    for (int j = 0; j < kBEAMS; ++j) {
      float best = -3.4e38f; int bi = 0;
      for (int i = 0; i < kBEAMS * kBEAMS; ++i)
        if (!used[i] && liks[i] > best) { best = liks[i]; bi = i; }
      used[bi] = true; ranked[j] = bi; rankedval[j] = best;
    }
  }
  __syncthreads();
  if (tid < kBEAMS) {
    int r = ranked[tid];
    int nt = choice[r];
    int oldb = r / kBEAMS;
    float nll = rankedval[tid];
    if (nt == kEND) nll += kMINLL;
    blls[b * kBEAMS + tid] = nll;
    const int* src = seq_in + (size_t)(b * kBEAMS + oldb) * kMS;
    int* dst = seq_out + (size_t)(b * kBEAMS + tid) * kMS;
    for (int i = 0; i < kMS; ++i) dst[i] = src[i];
    dst[ti + 1] = nt;
  }
}

// ---------------- init / finalize -----------------------------------------
__global__ void init_kernel(int* seqA, int* seqB, float* blls) {
  int idx = blockIdx.x * blockDim.x + threadIdx.x;
  if (idx < kBB * kMS) {
    int v = (idx % kMS == 0) ? kSTART : 0;
    seqA[idx] = v; seqB[idx] = v;
  }
  if (idx < kBB) blls[idx] = (idx % kBEAMS == 0) ? 0.f : kMINLL;
}

__global__ void finalize_kernel(const int* __restrict__ seq, const float* __restrict__ blls,
                                float* __restrict__ out) {
  int idx = blockIdx.x * blockDim.x + threadIdx.x;
  if (idx < kBB * kMS) ((int*)out)[idx] = seq[idx];     // int32 bits, first output
  else if (idx < kBB * kMS + kBB) out[idx] = blls[idx - kBB * kMS];
}

// ---------------- direct 3D conv, k=4, stride 2, pad 1, + bias + relu ------
__global__ void conv3d_kernel(const float* __restrict__ in, const float* __restrict__ w,
                              const float* __restrict__ bias, float* __restrict__ out,
                              int Ci, int Co, int Din, int Dout) {
  int idx = blockIdx.x * blockDim.x + threadIdx.x;
  int total = kBATCH * Co * Dout * Dout * Dout;
  if (idx >= total) return;
  int x = idx % Dout;
  int y = (idx / Dout) % Dout;
  int z = (idx / (Dout * Dout)) % Dout;
  int co = (idx / (Dout * Dout * Dout)) % Co;
  int n  =  idx / (Dout * Dout * Dout * Co);
  float acc = bias[co];
  for (int ci = 0; ci < Ci; ++ci) {
    const float* ip = in + ((size_t)n * Ci + ci) * Din * Din * Din;
    const float* wp = w + ((size_t)co * Ci + ci) * 64;
    for (int kz = 0; kz < 4; ++kz) {
      int iz = z * 2 + kz - 1; if (iz < 0 || iz >= Din) continue;
      for (int ky = 0; ky < 4; ++ky) {
        int iy = y * 2 + ky - 1; if (iy < 0 || iy >= Din) continue;
        for (int kx = 0; kx < 4; ++kx) {
          int ix = x * 2 + kx - 1; if (ix < 0 || ix >= Din) continue;
          acc += ip[((size_t)iz * Din + iy) * Din + ix] * wp[(kz * 4 + ky) * 4 + kx];
        }
      }
    }
  }
  out[idx] = acc > 0.f ? acc : 0.f;
}

// codes (8,256,2,2,2) -> bcodes (BB, MP, HD)  (identical across beams of a batch)
__global__ void codes_kernel(const float* __restrict__ x5, float* __restrict__ bcodes) {
  int idx = blockIdx.x * blockDim.x + threadIdx.x;
  if (idx >= kBB * kMP * kHD) return;
  int c = idx % kHD;
  int p = (idx / kHD) % kMP;
  int sb = idx / (kHD * kMP);
  int b = sb / kBEAMS;
  bcodes[idx] = x5[((size_t)b * kHD + c) * kMP + p];
}

// ---------------- workspace layout (bytes) ---------------------------------
static constexpr size_t alup(size_t x) { return (x + 255) & ~(size_t)255; }
static constexpr size_t O_BCODES = 0;
static constexpr size_t O_SEQA  = O_BCODES + alup((size_t)kBB * kMP * kHD * 4);
static constexpr size_t O_SEQB  = O_SEQA  + alup((size_t)kBB * kMS * 4);
static constexpr size_t O_BLLS  = O_SEQB  + alup((size_t)kBB * kMS * 4);
static constexpr size_t O_WQKV  = O_BLLS  + alup((size_t)kBB * 4);
static constexpr size_t O_WO    = O_WQKV  + alup((size_t)kNL * 768 * 256 * 2);
static constexpr size_t O_WFF1  = O_WO    + alup((size_t)kNL * 256 * 256 * 2);
static constexpr size_t O_WFF2  = O_WFF1  + alup((size_t)kNL * 1024 * 256 * 2);
static constexpr size_t O_HW1   = O_WFF2  + alup((size_t)kNL * 256 * 1024 * 2);
static constexpr size_t O_HW2   = O_HW1   + alup((size_t)256 * 256 * 2);
static constexpr size_t O_SCR   = O_HW2   + alup((size_t)kES * 256 * 2);
// conv ping-pong buffers (used before the decode loop only)
static constexpr size_t O_CONVP = O_SCR;
static constexpr size_t O_CONVQ = O_CONVP + alup((size_t)8 * 32 * 32 * 32 * 32 * 4);
// transformer per-step activations ALIAS the conv region (conv is done first)
static constexpr size_t O_X    = O_SCR;
static constexpr size_t O_H    = O_X    + alup((size_t)kBB * kT * kHD * 4);
static constexpr size_t O_QKV  = O_H    + alup((size_t)kBB * kT * kHD * 2);
static constexpr size_t O_ATT  = O_QKV  + alup((size_t)kBB * kT * 768 * 4);
static constexpr size_t O_FF1  = O_ATT  + alup((size_t)kBB * kT * kHD * 2);
static constexpr size_t O_HIN  = O_FF1  + alup((size_t)kBB * kT * 1024 * 2);
static constexpr size_t O_HMID = O_HIN  + alup((size_t)kHROWS * kHD * 2);
static constexpr size_t O_LOG  = O_HMID + alup((size_t)kHROWS * kHD * 2);
// total ws needed ~ 49 MB

extern "C" void kernel_launch(void* const* d_in, const int* in_sizes, int n_in,
                              void* d_out, int out_size, void* d_ws, size_t ws_size,
                              hipStream_t stream) {
  (void)in_sizes; (void)n_in; (void)out_size; (void)ws_size;

  const float* voxels = (const float*)d_in[0];
  const float* ck[5] = {(const float*)d_in[1], (const float*)d_in[3], (const float*)d_in[5],
                        (const float*)d_in[7], (const float*)d_in[9]};
  const float* cb[5] = {(const float*)d_in[2], (const float*)d_in[4], (const float*)d_in[6],
                        (const float*)d_in[8], (const float*)d_in[10]};
  const float* tok_emb = (const float*)d_in[11];
  const float* pos_emb = (const float*)d_in[12];
  const float* Wqkv  = (const float*)d_in[13];
  const float* Wo    = (const float*)d_in[14];
  const float* ln1_s = (const float*)d_in[15];
  const float* ln1_b = (const float*)d_in[16];
  const float* ln2_s = (const float*)d_in[17];
  const float* ln2_b = (const float*)d_in[18];
  const float* Wff1  = (const float*)d_in[19];
  const float* bff1  = (const float*)d_in[20];
  const float* Wff2  = (const float*)d_in[21];
  const float* bff2  = (const float*)d_in[22];
  const float* hW1   = (const float*)d_in[23];
  const float* hb1   = (const float*)d_in[24];
  const float* hW2   = (const float*)d_in[25];
  const float* hb2   = (const float*)d_in[26];

  char* ws = (char*)d_ws;
  float*     bcodes = (float*)(ws + O_BCODES);
  int*       seqA   = (int*)(ws + O_SEQA);
  int*       seqB   = (int*)(ws + O_SEQB);
  float*     blls   = (float*)(ws + O_BLLS);
  _Float16*  wqkvT  = (_Float16*)(ws + O_WQKV);
  _Float16*  woT    = (_Float16*)(ws + O_WO);
  _Float16*  wff1T  = (_Float16*)(ws + O_WFF1);
  _Float16*  wff2T  = (_Float16*)(ws + O_WFF2);
  _Float16*  hw1T   = (_Float16*)(ws + O_HW1);
  _Float16*  hw2T   = (_Float16*)(ws + O_HW2);
  float*     xbuf   = (float*)(ws + O_X);
  _Float16*  hbuf   = (_Float16*)(ws + O_H);
  float*     qkv    = (float*)(ws + O_QKV);
  _Float16*  attout = (_Float16*)(ws + O_ATT);
  _Float16*  ff1    = (_Float16*)(ws + O_FF1);
  _Float16*  hin    = (_Float16*)(ws + O_HIN);
  _Float16*  hmid   = (_Float16*)(ws + O_HMID);
  float*     logits = (float*)(ws + O_LOG);
  float*     convP  = (float*)(ws + O_CONVP);
  float*     convQ  = (float*)(ws + O_CONVQ);

  // ---- convert weights to f16, transposed (N x K) ----
  wt_f16_kernel<<<dim3(768, 1, kNL), 256, 0, stream>>>(Wqkv, wqkvT, 256, 768);
  wt_f16_kernel<<<dim3(256, 1, kNL), 256, 0, stream>>>(Wo,   woT,   256, 256);
  wt_f16_kernel<<<dim3(1024, 1, kNL), 256, 0, stream>>>(Wff1, wff1T, 256, 1024);
  wt_f16_kernel<<<dim3(1024, 1, kNL), 256, 0, stream>>>(Wff2, wff2T, 1024, 256);
  wt_f16_kernel<<<dim3(256, 1, 1), 256, 0, stream>>>(hW1, hw1T, 256, 256);
  wt_f16_kernel<<<dim3(64,  1, 1), 256, 0, stream>>>(hW2, hw2T, 256, 64);

  // ---- CNN encoder (ping-pong P/Q) ----
  {
    const int Ci[5]   = {1, 32, 64, 128, 256};
    const int Co[5]   = {32, 64, 128, 256, 256};
    const int Din[5]  = {64, 32, 16, 8, 4};
    const int Dout[5] = {32, 16, 8, 4, 2};
    float* bufs[2] = {convP, convQ};
    const float* cin = voxels;
    for (int i = 0; i < 5; ++i) {
      float* cout = bufs[i & 1];
      int total = kBATCH * Co[i] * Dout[i] * Dout[i] * Dout[i];
      conv3d_kernel<<<dim3((total + 255) / 256), 256, 0, stream>>>(
          cin, ck[i], cb[i], cout, Ci[i], Co[i], Din[i], Dout[i]);
      cin = cout;
    }
  }
  codes_kernel<<<dim3((kBB * kMP * kHD + 255) / 256), 256, 0, stream>>>(convP, bcodes);

  // ---- beam-search state init ----
  init_kernel<<<5, 256, 0, stream>>>(seqA, seqB, blls);

  // ---- decode loop: 31 serial steps ----
  int* cur = seqA;
  int* nxt = seqB;
  constexpr int ROWS = kBB * kT;   // 1600
  for (int ti = 0; ti < kMS - 1; ++ti) {
    embed_kernel<<<dim3(ROWS), 256, 0, stream>>>(bcodes, tok_emb, pos_emb, cur, xbuf);
    for (int l = 0; l < kNL; ++l) {
      ln_kernel<<<dim3(ROWS / 8), 256, 0, stream>>>(xbuf, ln1_s + l * kHD, ln1_b + l * kHD,
                                                    hbuf, ROWS);
      // qkv = h @ Wqkv[l]                                       (1600x768x256)
      gemm_wmma_kernel<false, false, false, false><<<dim3(24, 50), 32, 0, stream>>>(
          hbuf, wqkvT + (size_t)l * 768 * 256, nullptr, nullptr, qkv, nullptr, ROWS, 768, 256);
      attn_kernel<<<dim3(kBB * kNH), 64, 0, stream>>>(qkv, attout);
      // x += att @ Wo[l]                                        (1600x256x256)
      gemm_wmma_kernel<false, false, true, false><<<dim3(8, 50), 32, 0, stream>>>(
          attout, woT + (size_t)l * 256 * 256, nullptr, xbuf, xbuf, nullptr, ROWS, 256, 256);
      ln_kernel<<<dim3(ROWS / 8), 256, 0, stream>>>(xbuf, ln2_s + l * kHD, ln2_b + l * kHD,
                                                    hbuf, ROWS);
      // ff1 = relu(h @ Wff1[l] + bff1[l])                       (1600x1024x256)
      gemm_wmma_kernel<true, true, false, true><<<dim3(32, 50), 32, 0, stream>>>(
          hbuf, wff1T + (size_t)l * 1024 * 256, bff1 + l * 1024, nullptr, nullptr, ff1,
          ROWS, 1024, 256);
      // x += ff1 @ Wff2[l] + bff2[l]                            (1600x256x1024)
      gemm_wmma_kernel<true, false, true, false><<<dim3(8, 50), 32, 0, stream>>>(
          ff1, wff2T + (size_t)l * 256 * 1024, bff2 + l * kHD, xbuf, xbuf, nullptr,
          ROWS, 256, 1024);
    }
    // head on position MP+ti only (40 rows, padded to 64)
    headgather_kernel<<<dim3(kHROWS), 256, 0, stream>>>(xbuf, hin, ti);
    gemm_wmma_kernel<true, true, false, true><<<dim3(8, 2), 32, 0, stream>>>(
        hin, hw1T, hb1, nullptr, nullptr, hmid, kHROWS, 256, 256);
    gemm_wmma_kernel<true, false, false, false><<<dim3(2, 2), 32, 0, stream>>>(
        hmid, hw2T, hb2, nullptr, logits, nullptr, kHROWS, 64, 256);
    beam_kernel<<<dim3(kBATCH), 32, 0, stream>>>(logits, cur, nxt, blls, ti);
    int* tmp = cur; cur = nxt; nxt = tmp;
  }

  // outputs: bseqs (1280 int32, bit-copied) then blls (40 f32)
  finalize_kernel<<<dim3(6), 256, 0, stream>>>(cur, blls, (float*)d_out);
}
